// WorkingMemory_81003083202864
// MI455X (gfx1250) — matrix operand order repaired and checked by reading, not compiled, not run on hardware
//
#include <hip/hip_runtime.h>
#include <hip/hip_bf16.h>
#include <math.h>

typedef __attribute__((ext_vector_type(16))) __bf16 v16bf;
typedef __attribute__((ext_vector_type(8)))  float  v8f;
typedef __attribute__((ext_vector_type(4)))  float  f4;   // native ext-vector float4

#define NB 524288   // B
#define ND 64       // D
#define NS 4        // S

// Hardware tanh (V_TANH_F32 exists on gfx1250; TRANS-class op)
static __device__ __forceinline__ float fast_tanh(float x) {
#if __has_builtin(__builtin_amdgcn_tanhf)
    return __builtin_amdgcn_tanhf(x);
#else
    return tanhf(x);
#endif
}

// exp(x) via V_EXP_F32 (2^x)
static __device__ __forceinline__ float fast_exp(float x) {
#if __has_builtin(__builtin_amdgcn_exp2f)
    return __builtin_amdgcn_exp2f(x * 1.4426950408889634f);
#else
    return expf(x);
#endif
}

// float -> bf16 (RNE); native clang conversion lowers to HW cvt on gfx1250
static __device__ __forceinline__ __bf16 f2bf(float f) {
    return (__bf16)f;
}

// ---------------------------------------------------------------------------
// Kernel A: keys = tanh(slots @ Wk^T + bk); normalized softplus strengths;
// init scratch accumulators; pre-swizzle Wq into bf16 WMMA B-fragments.
// ---------------------------------------------------------------------------
__global__ void __launch_bounds__(256) prep_kernel(
    const float* __restrict__ slots, const float* __restrict__ ss,
    const float* __restrict__ Wk,    const float* __restrict__ bk,
    const float* __restrict__ Wq,
    float* __restrict__ ws_keys,     // [4*64]
    float* __restrict__ ws_str,      // [4]
    float* __restrict__ ws_acc,      // [4]
    int*   __restrict__ ws_lastw,    // [4]
    __bf16* __restrict__ ws_wqfrag)  // [8 frags][32 lanes][16] bf16
{
    int t = threadIdx.x;
    __shared__ float sp[NS];

    // keys (4x64 = exactly 256 threads)
    int s = t >> 6, d = t & 63;
    float acc = bk[d];
    #pragma unroll 8
    for (int k = 0; k < ND; ++k) acc += slots[s * ND + k] * Wk[d * ND + k];
    ws_keys[s * ND + d] = fast_tanh(acc);

    if (t < NS) {
        float x = ss[t];
        sp[t] = (x > 20.f) ? x : log1pf(expf(x));   // softplus
        ws_acc[t]   = 0.f;
        ws_lastw[t] = -1;
    }
    __syncthreads();
    if (t < NS) {
        float sum = sp[0] + sp[1] + sp[2] + sp[3];
        ws_str[t] = sp[t] / sum;
    }

    // Wq -> bf16 B-fragments for v_wmma_f32_16x16x32_bf16.
    // frag f = j*2+kk (N-tile j, K-chunk kk). For lane: n = lane%16, kh = lane/16,
// element e -> B[K][n] with K = kk*32 + kh*8 + e (e<8), kk*32 + 16 + kh*8 + (e-8) (e>=8), and B[K][n] = Wq[j*16+n][K].
    int f = t >> 5, lane = t & 31;
    int j = f >> 1, kk = f & 1;
    int n = lane & 15, kh = lane >> 4;
    const float* src = Wq + (size_t)(j * 16 + n) * ND + kk * 32 + kh * 8;
    __bf16* dst = ws_wqfrag + ((size_t)(f * 32 + lane)) * 16;
    #pragma unroll
    for (int e = 0; e < 16; ++e) dst[e] = f2bf(src[e < 8 ? e : e + 8]);
}

// ---------------------------------------------------------------------------
// Kernel B: main pass over items. WMMA projection + fused softmax/argmax +
// per-slot reductions (usage mean, last-writer).
// ---------------------------------------------------------------------------
__global__ void __launch_bounds__(256) main_kernel(
    const float* __restrict__ item,
    const float* __restrict__ bq,
    const float* __restrict__ ws_keys,
    const float* __restrict__ ws_str,
    const __bf16* __restrict__ ws_wqfrag,
    float* __restrict__ ws_acc,
    int*   __restrict__ ws_lastw,
    float* __restrict__ out_sel,     // [B] (float-cast of int slot index)
    float* __restrict__ out_sw)      // [B*4]
{
    __shared__ float qtile[8][16][ND];   // 32 KB: per-wave 16x64 query tile
    __shared__ float lkeys[NS][ND];      // 1 KB
    __shared__ float lstr[NS];
    __shared__ float lacc[NS];
    __shared__ int   llast[NS];

    int t = threadIdx.x;
    int wave = t >> 5, lane = t & 31;
    int block_row0 = blockIdx.x * 128;
    int row0 = block_row0 + wave * 16;
    int M = lane & 15, h = lane >> 4;

    if (t < NS) { lacc[t] = 0.f; llast[t] = -1; lstr[t] = ws_str[t]; }
    lkeys[t >> 6][t & 63] = ws_keys[t];   // 256 threads = 4x64

    // A-fragments (ISA 16-bit A layout): lane holds, per K-chunk, runs
    // K = kk*32 + h*8 + [0..8) for e=0..7 and K = kk*32 + 16 + h*8 + [0..8) for e=8..15.
    // Streaming read: non-temporal b128 loads.
    v16bf afrag[2];
    const float* irow = item + (size_t)(row0 + M) * ND;
    #pragma unroll
    for (int kk = 0; kk < 2; ++kk) {
        f4 a0 = __builtin_nontemporal_load((const f4*)(irow + kk * 32 + h * 8));
        f4 a1 = __builtin_nontemporal_load((const f4*)(irow + kk * 32 + h * 8 + 4));
        f4 a2 = __builtin_nontemporal_load((const f4*)(irow + kk * 32 + 16 + h * 8));
        f4 a3 = __builtin_nontemporal_load((const f4*)(irow + kk * 32 + 16 + h * 8 + 4));
        afrag[kk][0]  = f2bf(a0.x); afrag[kk][1]  = f2bf(a0.y);
        afrag[kk][2]  = f2bf(a0.z); afrag[kk][3]  = f2bf(a0.w);
        afrag[kk][4]  = f2bf(a1.x); afrag[kk][5]  = f2bf(a1.y);
        afrag[kk][6]  = f2bf(a1.z); afrag[kk][7]  = f2bf(a1.w);
        afrag[kk][8]  = f2bf(a2.x); afrag[kk][9]  = f2bf(a2.y);
        afrag[kk][10] = f2bf(a2.z); afrag[kk][11] = f2bf(a2.w);
        afrag[kk][12] = f2bf(a3.x); afrag[kk][13] = f2bf(a3.y);
        afrag[kk][14] = f2bf(a3.z); afrag[kk][15] = f2bf(a3.w);
    }

    // 4 N-tiles x (K=64 as 2x32): 8 WMMAs per wave-tile
    #pragma unroll
    for (int j = 0; j < 4; ++j) {
        v8f c = {0.f, 0.f, 0.f, 0.f, 0.f, 0.f, 0.f, 0.f};
        #pragma unroll
        for (int kk = 0; kk < 2; ++kk) {
            v16bf b = *(const v16bf*)(ws_wqfrag + ((size_t)((j * 2 + kk) * 32 + lane)) * 16);
            c = __builtin_amdgcn_wmma_f32_16x16x32_bf16(
                    false, afrag[kk], false, b, (short)0, c, false, false);
        }
        float bias = bq[j * 16 + (lane & 15)];
        #pragma unroll
        for (int v = 0; v < 8; ++v) {                 // C/D layout: row = v + 8*(lane/16)
            int row = v + 8 * h;
            qtile[wave][row][j * 16 + (lane & 15)] = fast_tanh(c[v] + bias);
        }
    }
    __syncthreads();

    // Per-row tail: sim (S=4 dots), softmax, strength weighting, argmax.
    if (t < 128) {
        int w = t >> 4, r = t & 15;
        int grow = block_row0 + t;
        const f4* q4 = (const f4*)qtile[w][r];

        float sim[NS];
        #pragma unroll
        for (int s2 = 0; s2 < NS; ++s2) {
            const f4* k4 = (const f4*)lkeys[s2];
            float a = 0.f;
            #pragma unroll
            for (int k = 0; k < ND / 4; ++k) {
                f4 qq = q4[k], kk2 = k4[k];
                a += qq.x * kk2.x + qq.y * kk2.y + qq.z * kk2.z + qq.w * kk2.w;
            }
            sim[s2] = a;
        }
        float m = fmaxf(fmaxf(sim[0], sim[1]), fmaxf(sim[2], sim[3]));
        float e[NS], esum = 0.f;
        #pragma unroll
        for (int s2 = 0; s2 < NS; ++s2) { e[s2] = fast_exp(sim[s2] - m); esum += e[s2]; }
        float inv = 1.f / esum;

        float p[NS], u[NS], usum = 0.f;
        #pragma unroll
        for (int s2 = 0; s2 < NS; ++s2) {
            p[s2] = e[s2] * inv;
            u[s2] = p[s2] * lstr[s2];
            usum += u[s2];
        }
        // slot_weights (pre-strength softmax): one b128 NT store per row
        f4 pw = {p[0], p[1], p[2], p[3]};
        __builtin_nontemporal_store(pw, (f4*)(out_sw + (size_t)grow * NS));

        float uinv = 1.f / usum;
        int best = 0; float bvv = u[0];
        #pragma unroll
        for (int s2 = 1; s2 < NS; ++s2) {
            if (u[s2] > bvv) { bvv = u[s2]; best = s2; }   // first-max semantics
        }
        __builtin_nontemporal_store((float)best, out_sel + grow);

        atomicMax(&llast[best], grow);
        #pragma unroll
        for (int s2 = 0; s2 < NS; ++s2) atomicAdd(&lacc[s2], u[s2] * uinv);
    }
    __syncthreads();

    if (t < NS) {
        atomicAdd(&ws_acc[t], lacc[t]);
        atomicMax(&ws_lastw[t], llast[t]);
    }
}

// ---------------------------------------------------------------------------
// Kernel C: gather new_slots by last writer, new_usage, out1 = tanh(mean@Wv^T+bv)
// ---------------------------------------------------------------------------
__global__ void __launch_bounds__(256) finalize_kernel(
    const float* __restrict__ item,  const float* __restrict__ slots,
    const float* __restrict__ usage, const float* __restrict__ Wv,
    const float* __restrict__ bvv,
    const float* __restrict__ ws_acc, const int* __restrict__ ws_lastw,
    float* __restrict__ out_newslots, float* __restrict__ out_newusage,
    float* __restrict__ ws_out1)
{
    __shared__ float ns[NS][ND];
    __shared__ float mean[ND];
    int t = threadIdx.x;
    int s = t >> 6, d = t & 63;

    int lw = ws_lastw[s];
    float v = (lw >= 0) ? item[(size_t)lw * ND + d] : slots[s * ND + d];
    ns[s][d] = v;
    out_newslots[s * ND + d] = v;

    if (t < NS)
        out_newusage[t] = usage[t] * 0.9f + ws_acc[t] * (1.0f / (float)NB);

    __syncthreads();
    if (t < ND) mean[t] = 0.25f * (ns[0][t] + ns[1][t] + ns[2][t] + ns[3][t]);
    __syncthreads();
    if (t < ND) {
        float a = bvv[t];
        #pragma unroll 8
        for (int k = 0; k < ND; ++k) a += mean[k] * Wv[t * ND + k];
        ws_out1[t] = fast_tanh(a);
    }
}

// ---------------------------------------------------------------------------
// Kernel D: broadcast out1[64] to output[B,64] (pure NT b128 store stream)
// ---------------------------------------------------------------------------
__global__ void __launch_bounds__(256) broadcast_kernel(
    const float* __restrict__ ws_out1, float* __restrict__ out)
{
    __shared__ float o[ND];
    if (threadIdx.x < ND) o[threadIdx.x] = ws_out1[threadIdx.x];
    __syncthreads();
    size_t idx = ((size_t)blockIdx.x * blockDim.x + threadIdx.x) * 4;
    int d = (int)(idx & 63);
    f4 val = {o[d], o[d + 1], o[d + 2], o[d + 3]};
    __builtin_nontemporal_store(val, (f4*)(out + idx));
}

// ---------------------------------------------------------------------------
extern "C" void kernel_launch(void* const* d_in, const int* in_sizes, int n_in,
                              void* d_out, int out_size, void* d_ws, size_t ws_size,
                              hipStream_t stream) {
    const float* item  = (const float*)d_in[0];   // [B,64]
    const float* slots = (const float*)d_in[1];   // [4,64]
    const float* ss    = (const float*)d_in[2];   // [4]
    const float* usage = (const float*)d_in[3];   // [4]
    const float* Wq    = (const float*)d_in[4];
    const float* bq    = (const float*)d_in[5];
    const float* Wk    = (const float*)d_in[6];
    const float* bk    = (const float*)d_in[7];
    const float* Wv    = (const float*)d_in[8];
    const float* bv    = (const float*)d_in[9];

    // scratch layout (byte offsets into d_ws)
    char* ws = (char*)d_ws;
    float*  ws_keys   = (float*)(ws + 0);        // 256 f32
    float*  ws_str    = (float*)(ws + 1024);     // 4 f32
    float*  ws_acc    = (float*)(ws + 1040);     // 4 f32
    int*    ws_lastw  = (int*)  (ws + 1056);     // 4 i32
    float*  ws_out1   = (float*)(ws + 1072);     // 64 f32
    __bf16* ws_wqfrag = (__bf16*)(ws + 4096);    // 4096 bf16 (8 KB)

    // d_out layout: output | new_slots | new_usage | selected_slot | slot_weights
    float* out        = (float*)d_out;
    float* out_ns     = out + (size_t)NB * ND;                       // +33554432
    float* out_nu     = out_ns + NS * ND;                            // +256
    float* out_sel    = out_nu + NS;                                 // +4
    float* out_sw     = out_sel + (size_t)NB;                        // +524288

    prep_kernel<<<1, 256, 0, stream>>>(slots, ss, Wk, bk, Wq,
                                       ws_keys, ws_str, ws_acc, ws_lastw, ws_wqfrag);

    main_kernel<<<NB / 128, 256, 0, stream>>>(item, bq, ws_keys, ws_str, ws_wqfrag,
                                              ws_acc, ws_lastw, out_sel, out_sw);

    finalize_kernel<<<1, 256, 0, stream>>>(item, slots, usage, Wv, bv,
                                           ws_acc, ws_lastw, out_ns, out_nu, ws_out1);

    broadcast_kernel<<<(NB * ND / 4) / 256, 256, 0, stream>>>(ws_out1, out);
}
